// CTSPd_Decoder_30150670418166
// MI455X (gfx1250) — matrix operand order repaired and checked by reading, not compiled, not run on hardware
//
#include <hip/hip_runtime.h>
#include <hip/hip_bf16.h>
#include <math.h>
#include <stdint.h>

#define HEAD_NUM 16
#define QKV_DIM  32
#define EMB      512
#define BATCH    32
#define POMO     256
#define PROBLEM  1024

typedef __attribute__((ext_vector_type(16))) _Float16 v16h;
typedef __attribute__((ext_vector_type(8)))  _Float16 v8h;
typedef __attribute__((ext_vector_type(8)))  float    v8f;
typedef __attribute__((ext_vector_type(4)))  unsigned int v4u;
typedef __attribute__((ext_vector_type(8)))  int      v8i;
typedef __attribute__((ext_vector_type(4)))  int      v4i;

// ---------------- WMMA fragment helpers (wave32, 16x16x32 f16) ----------------

// A fragment: 16x32 f16, row-major source. lane L: row = L&15,
// elems 0..7 -> K = (L>>4)*8 + 0..7 ; elems 8..15 -> K = 16 + (L>>4)*8 + 0..7
__device__ __forceinline__ v16h a_frag_f16(const _Float16* A, int lda, int k0) {
  const int lane = threadIdx.x & 31;
  const int hi   = lane >> 4;
  const _Float16* p = A + (size_t)(lane & 15) * lda + k0 + hi * 8;
  v8h lo = *(const v8h*)p;
  v8h hv = *(const v8h*)(p + 16);
  v16h r;
#pragma unroll
  for (int i = 0; i < 8; ++i) { r[i] = lo[i]; r[i + 8] = hv[i]; }
  return r;
}

// Same A fragment but source is fp32 (convert on the fly)
__device__ __forceinline__ v16h a_frag_f32(const float* A, int lda, int k0) {
  const int lane = threadIdx.x & 31;
  const int hi   = lane >> 4;
  const float* p = A + (size_t)(lane & 15) * lda + k0 + hi * 8;
  v16h r;
#pragma unroll
  for (int i = 0; i < 8; ++i) { r[i] = (_Float16)p[i]; r[i + 8] = (_Float16)p[i + 16]; }
  return r;
}

// B fragment: 32x16 f16, source stored N-major (Bt[n][k]). lane L: col n = L&15,
// elems 0..7 -> K = (L>>4)*8 + 0..7 ; elems 8..15 -> K = 16 + (L>>4)*8 + 0..7
__device__ __forceinline__ v16h b_frag_f16(const _Float16* Bt, int ldb, int k0) {
  const int lane = threadIdx.x & 31;
  const int hi   = lane >> 4;
  const _Float16* p = Bt + (size_t)(lane & 15) * ldb + k0 + hi * 8;
  v8h lo = *(const v8h*)p;
  v8h hv = *(const v8h*)(p + 16);
  v16h r;
#pragma unroll
  for (int i = 0; i < 8; ++i) { r[i] = lo[i]; r[i + 8] = hv[i]; }
  return r;
}

__device__ __forceinline__ v8f wmma16(v16h a, v16h b, v8f c) {
  return __builtin_amdgcn_wmma_f32_16x16x32_f16(false, a, false, b, (short)0, c,
                                                false, false);
}

// C store (f16): VGPR r -> row = (lane>>4)*8 + r, col = lane&15
__device__ __forceinline__ void store_c_f16(_Float16* C, int ldc, v8f acc) {
  const int lane = threadIdx.x & 31;
  const int n  = lane & 15;
  const int hi = lane >> 4;
#pragma unroll
  for (int r = 0; r < 8; ++r)
    C[(size_t)(hi * 8 + r) * ldc + n] = (_Float16)acc[r];
}

// ---------------- Tensor Data Mover staging (gfx1250 TDM) ----------------
// Stage a dense [rows x elems] f16 tile (row stride `stride` elems in global)
// into contiguous LDS via TENSOR_LOAD_TO_LDS. Descriptor layout per
// CDNA5 ISA (2-D tensor, groups 2/3 zero).
__device__ __forceinline__ void stage_tile_f16(_Float16* lds, const _Float16* g,
                                               unsigned rows, unsigned elems,
                                               unsigned stride) {
#if __has_builtin(__builtin_amdgcn_tensor_load_to_lds)
  if (threadIdx.x < 32) {  // one wave issues the DMA
    unsigned long long ga = (unsigned long long)(uintptr_t)g;
    unsigned lds_off = (unsigned)(uintptr_t)lds;  // low 32b = LDS byte offset
    v4u g0;
    g0[0] = 1u;                                   // count=1 (valid), user mode
    g0[1] = lds_off;                              // lds_addr
    g0[2] = (unsigned)ga;                         // global_addr[31:0]
    g0[3] = (unsigned)((ga >> 32) & 0x01FFFFFFu)  // global_addr[56:32]
            | 0x80000000u;                        // type=2 ("image")
    v8i g1;
    g1[0] = (int)(1u << 16);                      // data_size=1 (2 bytes/elem)
    g1[1] = (int)((elems & 0xFFFFu) << 16);       // tensor_dim0[15:0]
    g1[2] = (int)(((elems >> 16) & 0xFFFFu) |     // tensor_dim0[31:16]
                  ((rows & 0xFFFFu) << 16));      // tensor_dim1[15:0]
    g1[3] = (int)(((rows >> 16) & 0xFFFFu) |      // tensor_dim1[31:16]
                  ((elems & 0xFFFFu) << 16));     // tile_dim0
    g1[4] = (int)(rows & 0xFFFFu);                // tile_dim1 (tile_dim2=0)
    g1[5] = (int)stride;                          // tensor_dim0_stride[31:0]
    g1[6] = 0;                                    // stride[47:32] | dim1_stride lo
    g1[7] = 0;
    v4i z4 = {};
#if defined(__clang_major__) && (__clang_major__ >= 23)
    v8i z8 = {};
    __builtin_amdgcn_tensor_load_to_lds(g0, g1, z4, z4, z8, 0);
#else
    __builtin_amdgcn_tensor_load_to_lds(g0, g1, z4, z4, 0);
#endif
    __builtin_amdgcn_s_wait_tensorcnt(0);
  }
#else
  for (unsigned i = threadIdx.x; i < rows * elems; i += blockDim.x)
    lds[i] = g[(size_t)(i / elems) * stride + (i % elems)];
#endif
  __syncthreads();
}

// ---------------- elementwise / layout prep ----------------

__global__ void k_cvt_f32_to_f16(const float* __restrict__ src,
                                 _Float16* __restrict__ dst, int n) {
  int i = blockIdx.x * blockDim.x + threadIdx.x;
  if (i < n) dst[i] = (_Float16)src[i];
}

// Wt[n][k] = (f16) W[k][n]   (W is [K][N] row-major)
__global__ void k_transpose_w(const float* __restrict__ W,
                              _Float16* __restrict__ Wt, int K, int N) {
  int i = blockIdx.x * blockDim.x + threadIdx.x;
  if (i < K * N) {
    int k = i / N, n = i % N;
    Wt[(size_t)n * K + k] = (_Float16)W[i];
  }
}

// ---------------- projection GEMMs ----------------
// Block: 256 thr = 8 waves -> 64x32 C tile (wave w: row sub (w>>1), col sub (w&1)).
// 32x512 f16 weight panel (32KB) staged once per block via TDM, reused 4x.

__global__ __launch_bounds__(256)
void k_proj_rowmajor(const float* __restrict__ A, const _Float16* __restrict__ Wt,
                     _Float16* __restrict__ C) {
  __shared__ _Float16 s_b[32 * EMB];  // 32 KB weight panel
  const int wave = threadIdx.x >> 5;
  const int row0 = blockIdx.x * 64 + (wave >> 1) * 16;
  const int col0 = blockIdx.y * 32 + (wave & 1) * 16;
  stage_tile_f16(s_b, Wt + (size_t)(blockIdx.y * 32) * EMB, 32, EMB, EMB);

  const float*    Ab = A + (size_t)row0 * EMB;
  const _Float16* Bb = s_b + (size_t)((wave & 1) * 16) * EMB;
  v8f acc = {};
  for (int k0 = 0; k0 < EMB; k0 += 32)
    acc = wmma16(a_frag_f32(Ab, EMB, k0), b_frag_f16(Bb, EMB, k0), acc);
  store_c_f16(C + (size_t)row0 * EMB + col0, EMB, acc);
}

// V projection with per-head transposed store: Vt[b][h][d][m]
__global__ __launch_bounds__(256)
void k_proj_vt(const float* __restrict__ A, const _Float16* __restrict__ Wt,
               _Float16* __restrict__ Vt) {
  __shared__ _Float16 s_b[32 * EMB];
  const int wave  = threadIdx.x >> 5;
  const int mrow0 = blockIdx.x * 64 + (wave >> 1) * 16;  // b*1024 + m
  const int col0  = blockIdx.y * 32 + (wave & 1) * 16;   // n in [0,512)
  stage_tile_f16(s_b, Wt + (size_t)(blockIdx.y * 32) * EMB, 32, EMB, EMB);

  const float*    Ab = A + (size_t)mrow0 * EMB;
  const _Float16* Bb = s_b + (size_t)((wave & 1) * 16) * EMB;
  v8f acc = {};
  for (int k0 = 0; k0 < EMB; k0 += 32)
    acc = wmma16(a_frag_f32(Ab, EMB, k0), b_frag_f16(Bb, EMB, k0), acc);

  const int lane = threadIdx.x & 31;
  const int hi   = lane >> 4;
  const int n    = col0 + (lane & 15);
  const int h = n >> 5, d = n & 31;
  const int b = mrow0 >> 10, m0 = (mrow0 & 1023) + hi * 8;
  _Float16* dst = Vt + (((size_t)b * HEAD_NUM + h) * QKV_DIM + d) * PROBLEM + m0;
#pragma unroll
  for (int r = 0; r < 8; ++r) dst[r] = (_Float16)acc[r];
}

// Q = q1 @ Wq_first + qlast @ Wq_last (dual accumulate, B from global)
__global__ __launch_bounds__(256)
void k_proj_q(const float* __restrict__ A1, const float* __restrict__ A2,
              const _Float16* __restrict__ W1t, const _Float16* __restrict__ W2t,
              _Float16* __restrict__ Q) {
  const int wave = threadIdx.x >> 5;
  const int row0 = blockIdx.x * 64 + (wave >> 1) * 16;
  const int col0 = blockIdx.y * 32 + (wave & 1) * 16;
  const float*    A1b = A1  + (size_t)row0 * EMB;
  const float*    A2b = A2  + (size_t)row0 * EMB;
  const _Float16* B1b = W1t + (size_t)col0 * EMB;
  const _Float16* B2b = W2t + (size_t)col0 * EMB;
  v8f acc = {};
  for (int k0 = 0; k0 < EMB; k0 += 32) {
    __builtin_prefetch(B1b + (size_t)(threadIdx.x & 15) * EMB + k0 + 64, 0, 1);
    acc = wmma16(a_frag_f32(A1b, EMB, k0), b_frag_f16(B1b, EMB, k0), acc);
    acc = wmma16(a_frag_f32(A2b, EMB, k0), b_frag_f16(B2b, EMB, k0), acc);
  }
  store_c_f16(Q + (size_t)row0 * EMB + col0, EMB, acc);
}

// ---------------- multi-head attention core ----------------
// One block per (b, h, 16-row q tile). Scores staged in LDS, softmax, then W@V.
__global__ __launch_bounds__(256)
void k_attention(const _Float16* __restrict__ Q, const _Float16* __restrict__ K,
                 const _Float16* __restrict__ Vt, const float* __restrict__ ninf,
                 _Float16* __restrict__ OC) {
  __shared__ _Float16 s_w[16][PROBLEM];   // 32 KB
  __shared__ float    s_red[16][16];
  __shared__ float    s_max[16];
  __shared__ float    s_inv[16];

  const int qt = blockIdx.x, h = blockIdx.y, b = blockIdx.z;
  const int q0 = qt * 16;
  const int wave = threadIdx.x >> 5;
  const int lane = threadIdx.x & 31;
  const int hi   = lane >> 4;

  const _Float16* Qb = Q + ((size_t)(b * POMO + q0)) * EMB + h * QKV_DIM;
  const _Float16* Kb = K + ((size_t)b * PROBLEM) * EMB + h * QKV_DIM;
  const float* maskb = ninf + ((size_t)(b * POMO + q0)) * PROBLEM;

  const v16h qa = a_frag_f16(Qb, EMB, 0);   // K=32 covers full head dim
  const float scale = 0.17677669529663687f; // 1/sqrt(32)
#pragma unroll
  for (int t = 0; t < 8; ++t) {
    const int col0 = (wave * 8 + t) * 16;
    v16h kb = b_frag_f16(Kb + (size_t)col0 * EMB, EMB, 0);
    v8f acc = {};
    acc = wmma16(qa, kb, acc);
    const int n = col0 + (lane & 15);
#pragma unroll
    for (int r = 0; r < 8; ++r) {
      const int p = hi * 8 + r;
      s_w[p][n] = (_Float16)(acc[r] * scale + maskb[(size_t)p * PROBLEM + n]);
    }
  }
  __syncthreads();

  // block softmax over 1024 per row: row = tid&15, 16 segments of 64
  const int row = threadIdx.x & 15;
  const int seg = threadIdx.x >> 4;
  float mx = -3.0e38f;
  for (int c = seg * 64; c < seg * 64 + 64; ++c) mx = fmaxf(mx, (float)s_w[row][c]);
  s_red[row][seg] = mx;
  __syncthreads();
  if (threadIdx.x < 16) {
    float m2 = -3.0e38f;
    for (int i = 0; i < 16; ++i) m2 = fmaxf(m2, s_red[threadIdx.x][i]);
    s_max[threadIdx.x] = m2;
  }
  __syncthreads();
  const float rm = s_max[row];
  float sm = 0.f;
  for (int c = seg * 64; c < seg * 64 + 64; ++c) sm += __expf((float)s_w[row][c] - rm);
  s_red[row][seg] = sm;
  __syncthreads();
  if (threadIdx.x < 16) {
    float tot = 0.f;
    for (int i = 0; i < 16; ++i) tot += s_red[threadIdx.x][i];
    s_inv[threadIdx.x] = 1.0f / tot;
  }
  __syncthreads();
  const float inv = s_inv[row];
  for (int c = seg * 64; c < seg * 64 + 64; ++c)
    s_w[row][c] = (_Float16)(__expf((float)s_w[row][c] - rm) * inv);
  __syncthreads();

  // out = weights @ V : 16x32 tile; waves 0,1 (A from LDS -> ds_load, B = Vt)
  if (wave < 2) {
    const int col0 = wave * 16;  // d
    const _Float16* Vb =
        Vt + (((size_t)b * HEAD_NUM + h) * QKV_DIM + col0) * PROBLEM;
    v8f acc = {};
    for (int k0 = 0; k0 < PROBLEM; k0 += 32) {
      v16h aw = a_frag_f16(&s_w[0][0], PROBLEM, k0);
      v16h bv = b_frag_f16(Vb, PROBLEM, k0);
      acc = wmma16(aw, bv, acc);
    }
    store_c_f16(OC + ((size_t)(b * POMO + q0)) * EMB + h * QKV_DIM + col0, EMB, acc);
  }
}

// ---------------- combine GEMM (+bias), weight panel via TDM ----------------
__global__ __launch_bounds__(256)
void k_combine(const _Float16* __restrict__ OC, const _Float16* __restrict__ Wct,
               const float* __restrict__ bias, _Float16* __restrict__ MH) {
  __shared__ _Float16 s_b[32 * EMB];  // 32 KB
  const int wave = threadIdx.x >> 5;
  const int row0 = blockIdx.x * 64 + (wave >> 1) * 16;
  const int col0 = blockIdx.y * 32 + (wave & 1) * 16;
  stage_tile_f16(s_b, Wct + (size_t)(blockIdx.y * 32) * EMB, 32, EMB, EMB);

  const _Float16* Ab = OC + (size_t)row0 * EMB;
  const _Float16* Bb = s_b + (size_t)((wave & 1) * 16) * EMB;
  v8f acc = {};
  for (int k0 = 0; k0 < EMB; k0 += 32)
    acc = wmma16(a_frag_f16(Ab, EMB, k0), b_frag_f16(Bb, EMB, k0), acc);
  const float bv = bias[col0 + (threadIdx.x & 15)];
#pragma unroll
  for (int r = 0; r < 8; ++r) acc[r] += bv;
  store_c_f16(MH + (size_t)row0 * EMB + col0, EMB, acc);
}

// ---------------- score2 = clip(mh @ nodes^T) + mask ----------------
// block = 16 rows x full 1024 cols; 8 waves x 8 tiles, K=512.
// The shared 16x512 A tile (16KB) is staged once via TDM and reused by all
// waves across all K steps (8x global-traffic reduction on A).
__global__ __launch_bounds__(256)
void k_score2(const _Float16* __restrict__ MH, const _Float16* __restrict__ N16,
              const float* __restrict__ ninf, float* __restrict__ out) {
  __shared__ _Float16 s_a[16 * EMB];  // 16 KB
  const int pt = blockIdx.x;          // 0..511 (b*16 + qtile)
  const int b  = pt >> 4;
  const int wave = threadIdx.x >> 5;
  const int lane = threadIdx.x & 31;
  const int hi   = lane >> 4;
  stage_tile_f16(s_a, MH + (size_t)pt * 16 * EMB, 16, EMB, EMB);

  const _Float16* Nb = N16 + (size_t)b * PROBLEM * EMB;
  const float* maskb = ninf + (size_t)pt * 16 * PROBLEM;
  float* ob = out + (size_t)pt * 16 * PROBLEM;

  v8f acc[8] = {};
  for (int k0 = 0; k0 < EMB; k0 += 32) {
    v16h a = a_frag_f16(s_a, EMB, k0);  // ds_load from staged tile
#pragma unroll
    for (int t = 0; t < 8; ++t) {
      v16h bf = b_frag_f16(Nb + (size_t)((wave * 8 + t) * 16) * EMB, EMB, k0);
      acc[t] = wmma16(a, bf, acc[t]);
    }
  }
  const float inv_s = 1.0f / 22.627416997969522f;  // 1/sqrt(512)
#pragma unroll
  for (int t = 0; t < 8; ++t) {
    const int n = (wave * 8 + t) * 16 + (lane & 15);
#pragma unroll
    for (int r = 0; r < 8; ++r) {
      const int p = hi * 8 + r;
      ob[(size_t)p * PROBLEM + n] =
          10.0f * tanhf(acc[t][r] * inv_s) + maskb[(size_t)p * PROBLEM + n];
    }
  }
}

// ---------------- final row softmax (in-place on d_out) ----------------
__global__ __launch_bounds__(256)
void k_softmax_row(float* __restrict__ data) {
  __shared__ float red[256];
  float* row = data + (size_t)blockIdx.x * PROBLEM;
  const int t = threadIdx.x;
  float v0 = row[t], v1 = row[t + 256], v2 = row[t + 512], v3 = row[t + 768];
  red[t] = fmaxf(fmaxf(v0, v1), fmaxf(v2, v3));
  __syncthreads();
  for (int s = 128; s > 0; s >>= 1) {
    if (t < s) red[t] = fmaxf(red[t], red[t + s]);
    __syncthreads();
  }
  const float mx = red[0];
  __syncthreads();
  v0 = __expf(v0 - mx); v1 = __expf(v1 - mx);
  v2 = __expf(v2 - mx); v3 = __expf(v3 - mx);
  red[t] = v0 + v1 + v2 + v3;
  __syncthreads();
  for (int s = 128; s > 0; s >>= 1) {
    if (t < s) red[t] += red[t + s];
    __syncthreads();
  }
  const float inv = 1.0f / red[0];
  row[t] = v0 * inv; row[t + 256] = v1 * inv;
  row[t + 512] = v2 * inv; row[t + 768] = v3 * inv;
}

// ---------------- host launcher ----------------
extern "C" void kernel_launch(void* const* d_in, const int* in_sizes, int n_in,
                              void* d_out, int out_size, void* d_ws, size_t ws_size,
                              hipStream_t stream) {
  (void)in_sizes; (void)n_in; (void)out_size; (void)ws_size;
  const float* nodes = (const float*)d_in[0];
  const float* q1    = (const float*)d_in[1];
  const float* qlast = (const float*)d_in[2];
  const float* ninf  = (const float*)d_in[3];
  const float* Wq1   = (const float*)d_in[4];
  const float* Wq2   = (const float*)d_in[5];
  const float* Wk    = (const float*)d_in[6];
  const float* Wv    = (const float*)d_in[7];
  const float* Wc    = (const float*)d_in[8];
  const float* bc    = (const float*)d_in[9];
  float* out = (float*)d_out;

  char* ws = (char*)d_ws;
  size_t off = 0;
  auto alloc = [&](size_t bytes) {
    void* p = ws + off;
    off += (bytes + 255) & ~(size_t)255;
    return p;
  };
  _Float16* nodes16 = (_Float16*)alloc((size_t)BATCH * PROBLEM * EMB * 2);
  _Float16* Wq1t = (_Float16*)alloc((size_t)EMB * EMB * 2);
  _Float16* Wq2t = (_Float16*)alloc((size_t)EMB * EMB * 2);
  _Float16* Wkt  = (_Float16*)alloc((size_t)EMB * EMB * 2);
  _Float16* Wvt  = (_Float16*)alloc((size_t)EMB * EMB * 2);
  _Float16* Wct  = (_Float16*)alloc((size_t)EMB * EMB * 2);
  _Float16* K16  = (_Float16*)alloc((size_t)BATCH * PROBLEM * EMB * 2);
  _Float16* Vt16 = (_Float16*)alloc((size_t)BATCH * PROBLEM * EMB * 2);
  _Float16* Q16  = (_Float16*)alloc((size_t)BATCH * POMO * EMB * 2);
  _Float16* OC16 = (_Float16*)alloc((size_t)BATCH * POMO * EMB * 2);
  _Float16* MH16 = (_Float16*)alloc((size_t)BATCH * POMO * EMB * 2);

  // 1) layout prep
  {
    int n = BATCH * PROBLEM * EMB;
    k_cvt_f32_to_f16<<<(n + 255) / 256, 256, 0, stream>>>(nodes, nodes16, n);
    int nw = EMB * EMB;
    dim3 g((nw + 255) / 256);
    k_transpose_w<<<g, 256, 0, stream>>>(Wq1, Wq1t, EMB, EMB);
    k_transpose_w<<<g, 256, 0, stream>>>(Wq2, Wq2t, EMB, EMB);
    k_transpose_w<<<g, 256, 0, stream>>>(Wk,  Wkt,  EMB, EMB);
    k_transpose_w<<<g, 256, 0, stream>>>(Wv,  Wvt,  EMB, EMB);
    k_transpose_w<<<g, 256, 0, stream>>>(Wc,  Wct,  EMB, EMB);
  }
  // 2) projections
  k_proj_rowmajor<<<dim3((BATCH * PROBLEM) / 64, EMB / 32), 256, 0, stream>>>(
      nodes, Wkt, K16);
  k_proj_vt<<<dim3((BATCH * PROBLEM) / 64, EMB / 32), 256, 0, stream>>>(
      nodes, Wvt, Vt16);
  k_proj_q<<<dim3((BATCH * POMO) / 64, EMB / 32), 256, 0, stream>>>(
      q1, qlast, Wq1t, Wq2t, Q16);
  // 3) attention
  k_attention<<<dim3(POMO / 16, HEAD_NUM, BATCH), 256, 0, stream>>>(
      Q16, K16, Vt16, ninf, OC16);
  // 4) combine
  k_combine<<<dim3((BATCH * POMO) / 64, EMB / 32), 256, 0, stream>>>(
      OC16, Wct, bc, MH16);
  // 5) single-head score + clip (writes d_out), then softmax in place
  k_score2<<<dim3(BATCH * POMO / 16), 256, 0, stream>>>(MH16, nodes16, ninf, out);
  k_softmax_row<<<dim3(BATCH * POMO), 256, 0, stream>>>(out);
}
